// PhysicsInformedThimble_29283087024431
// MI455X (gfx1250) — matrix operand order repaired and checked by reading, not compiled, not run on hardware
//
#include <hip/hip_runtime.h>

#define LAT  32
#define NB   16
#define NCH  64
#define SPAT 32768            /* 32^3 */
#define VOL  1048576          /* 32^4 */

typedef __attribute__((ext_vector_type(16))) __bf16 v16bf;
typedef __attribute__((ext_vector_type(8)))  __bf16 v8bf;
typedef __attribute__((ext_vector_type(8)))  float  v8f;
typedef unsigned int u32x4 __attribute__((ext_vector_type(4)));
typedef int          i32x4 __attribute__((ext_vector_type(4)));
typedef int          i32x8 __attribute__((ext_vector_type(8)));

// ---- Tensor Data Mover availability / arity probe --------------------------
#if defined(__has_builtin)
#  if __has_builtin(__builtin_amdgcn_tensor_load_to_lds) && \
      __has_builtin(__builtin_amdgcn_s_wait_tensorcnt)
#    define USE_TDM 1
#  endif
#endif
#ifndef USE_TDM
#  define USE_TDM 0
#endif
#if __has_include(<hip/amd_detail/amd_gfx1250_TDM.h>)
#  define TDM_ARGS6 1          /* therock-10.0 headers -> 6-arg builtin */
#else
#  define TDM_ARGS6 0          /* ROCm 7.2 -> 5-arg builtin */
#endif

__device__ __forceinline__ float silu_f(float v) { return v / (1.f + __expf(-v)); }
// Fast SiLU: v_exp_f32 + v_rcp_f32 instead of the IEEE divide sequence.
__device__ __forceinline__ float silu_fast(float v) {
  return v * __builtin_amdgcn_rcpf(1.f + __expf(-v));
}
// Fast tanh: (e^2x - 1) * rcp(e^2x + 1).
__device__ __forceinline__ float tanh_fast(float x) {
  float e = __expf(2.f * x);
  return (e - 1.f) * __builtin_amdgcn_rcpf(e + 1.f);
}

// ---------------------------------------------------------------------------
// Fully-unrolled 32-point radix-2 complex FFT in registers.
// ---------------------------------------------------------------------------
__device__ __forceinline__ void fft32(float re[32], float im[32], float sign) {
#pragma unroll
  for (int i = 0; i < 32; ++i) {
    int r = ((i & 1) << 4) | ((i & 2) << 2) | (i & 4) | ((i & 8) >> 2) | ((i & 16) >> 4);
    if (r > i) {
      float t = re[i]; re[i] = re[r]; re[r] = t;
      t = im[i]; im[i] = im[r]; im[r] = t;
    }
  }
#pragma unroll
  for (int len = 2; len <= 32; len <<= 1) {
    float ang = sign * 6.28318530717958647692f / (float)len;
#pragma unroll
    for (int i = 0; i < 32; i += len) {
#pragma unroll
      for (int j = 0; j < (len >> 1); ++j) {
        float s, c;
        __sincosf(ang * (float)j, &s, &c);
        int a = i + j, b2 = a + (len >> 1);
        float tr = re[b2] * c - im[b2] * s;
        float ti = re[b2] * s + im[b2] * c;
        re[b2] = re[a] - tr; im[b2] = im[a] - ti;
        re[a] += tr;         im[a] += ti;
      }
    }
  }
}

__global__ __launch_bounds__(256)
void fft_pass_kernel(const float* __restrict__ src, float* __restrict__ R,
                     float* __restrict__ I, int sh, int firstPass, float sign) {
  int lid = blockIdx.x * 256 + threadIdx.x;     // NB*SPAT lines, grid exact
  int b = lid >> 15;
  int l = lid & (SPAT - 1);
  int stride = 1 << sh;
  long long base = (long long)b * VOL + ((long long)(l >> sh) << (sh + 5)) + (l & (stride - 1));
  float re[32], im[32];
  if (firstPass) {
#pragma unroll
    for (int j = 0; j < 32; ++j) { re[j] = src[base + ((long long)j << sh)]; im[j] = 0.f; }
  } else {
#pragma unroll
    for (int j = 0; j < 32; ++j) {
      long long idx = base + ((long long)j << sh);
      re[j] = R[idx]; im[j] = I[idx];
    }
  }
  fft32(re, im, sign);
#pragma unroll
  for (int j = 0; j < 32; ++j) {
    long long idx = base + ((long long)j << sh);
    R[idx] = re[j]; I[idx] = im[j];
  }
}

__global__ __launch_bounds__(256)
void spectral_scale_kernel(float* __restrict__ R, float* __restrict__ I,
                           const float* __restrict__ p_vec) {
  int i = blockIdx.x * 256 + threadIdx.x;      // NB*VOL, grid exact
  float m2 = p_vec[0], kin = p_vec[1];
  int s = i & (VOL - 1);
  const float w = 0.19634954084936207f;        // 2*pi/32
  float e0 = 2.f * (1.f - __cosf(w * (float)((s >> 15) & 31)));
  float e1 = 2.f * (1.f - __cosf(w * (float)((s >> 10) & 31)));
  float e2 = 2.f * (1.f - __cosf(w * (float)((s >> 5) & 31)));
  float e3 = 2.f * (1.f - __cosf(w * (float)(s & 31)));
  float kern = fmaxf(kin * (e0 + e1 + e2 + e3) + m2, 1e-8f);
  float sc = rsqrtf(kern) * (1.f / 1048576.f);
  R[i] *= sc; I[i] *= sc;
}

// ---------------------------------------------------------------------------
__global__ void zero_kernel(float* p, int n) {
  int i = blockIdx.x * blockDim.x + threadIdx.x;
  if (i < n) p[i] = 0.f;
}

__global__ __launch_bounds__(256)
void logdet_const_kernel(const float* __restrict__ p_vec, float* __restrict__ scal) {
  float m2 = p_vec[0], kin = p_vec[1];
  const float w = 0.19634954084936207f;
  float local = 0.f;
  int gsz = gridDim.x * blockDim.x;
  for (int i = blockIdx.x * blockDim.x + threadIdx.x; i < VOL; i += gsz) {
    float e0 = 2.f * (1.f - __cosf(w * (float)((i >> 15) & 31)));
    float e1 = 2.f * (1.f - __cosf(w * (float)((i >> 10) & 31)));
    float e2 = 2.f * (1.f - __cosf(w * (float)((i >> 5) & 31)));
    float e3 = 2.f * (1.f - __cosf(w * (float)(i & 31)));
    float kern = fmaxf(kin * (e0 + e1 + e2 + e3) + m2, 1e-8f);
    local += -0.5f * __logf(kern);
  }
  __shared__ float red[256];
  red[threadIdx.x] = local; __syncthreads();
  for (int st = 128; st > 0; st >>= 1) {
    if (threadIdx.x < st) red[threadIdx.x] += red[threadIdx.x + st];
    __syncthreads();
  }
  if (threadIdx.x == 0) atomicAdd(scal, red[0]);
}

__global__ void logdet_bcast_kernel(const float* __restrict__ s, float* __restrict__ ld) {
  if (threadIdx.x < NB) ld[threadIdx.x] = s[0];
}

__global__ void param_mlp_kernel(const float* pv, const float* w1, const float* b1,
                                 const float* w2, const float* b2,
                                 const float* w3, const float* b3,
                                 const float* ppw, const float* ppb, float* pbias) {
  __shared__ float p[8], h1[32], h2[32], pe[8];
  int t = threadIdx.x;                         // 32 threads
  if (t < 8) p[t] = pv[t];
  __syncthreads();
  float a = b1[t];
  for (int c = 0; c < 8; ++c) a += w1[t * 8 + c] * p[c];
  h1[t] = silu_f(a);
  __syncthreads();
  a = b2[t];
  for (int c = 0; c < 32; ++c) a += w2[t * 32 + c] * h1[c];
  h2[t] = silu_f(a);
  __syncthreads();
  if (t < 8) {
    a = b3[t];
    for (int c = 0; c < 32; ++c) a += w3[t * 32 + c] * h2[c];
    pe[t] = a + p[t];
  }
  __syncthreads();
  a = ppb[t];
  for (int c = 0; c < 8; ++c) a += ppw[t * 8 + c] * pe[c];
  pbias[t] = a;
}

// Re-layout conv weights to tap-major bf16: dst[o][tap*Cin + c] = w[o][c][tap].
__global__ void prep_weights_kernel(const float* __restrict__ w, __bf16* __restrict__ dst,
                                    int Cin) {
  int Ktot = Cin * 27;
  int i = blockIdx.x * blockDim.x + threadIdx.x;
  if (i >= 64 * Ktot) return;
  int o = i / Ktot;
  int k = i - o * Ktot;
  int tap = k / Cin;
  int c   = k - tap * Cin;
  dst[i] = (__bf16)w[(o * Cin + c) * 27 + tap];
}

// GroupNorm statistics: one block per (batch, group) -> mu, rstd.
__global__ __launch_bounds__(256)
void gn_stats_kernel(const float* __restrict__ x, float* __restrict__ stats) {
  int bg = blockIdx.x;                         // b*8+g, 128 blocks
  const float* base = x + ((long long)(bg >> 3) * NCH + (bg & 7) * 8) * SPAT;
  float s = 0.f, ss = 0.f;
  for (int i = threadIdx.x; i < 8 * SPAT; i += 256) {
    float v = base[i];
    s += v; ss += v * v;
  }
  __shared__ float r0[256], r1[256];
  r0[threadIdx.x] = s; r1[threadIdx.x] = ss;
  __syncthreads();
  for (int st = 128; st > 0; st >>= 1) {
    if (threadIdx.x < st) { r0[threadIdx.x] += r0[threadIdx.x + st]; r1[threadIdx.x] += r1[threadIdx.x + st]; }
    __syncthreads();
  }
  if (threadIdx.x == 0) {
    float mu  = r0[0] * (1.f / 262144.f);
    float var = r1[0] * (1.f / 262144.f) - mu * mu;
    stats[bg * 2 + 0] = mu;
    stats[bg * 2 + 1] = rsqrtf(var + 1e-5f);
  }
}

// ---------------------------------------------------------------------------
// Implicit-GEMM 3x3x3 circular conv via V_WMMA_F32_16X16X32_BF16.
// Block = 128 threads = 4 waves; wave w owns out-channel tile [16w,16w+16).
// N-tile = full 32-wide x row (two 16-wide WMMA N tiles per wave, shared A).
// K = Cin*27 in TAP-MAJOR order (k = tap*Cin + c), chunked by 32.
// A-tiles (weights) streamed to LDS by the Tensor Data Mover each K chunk.
// Gather: thread owns (n = tid&31, 8 contiguous k) -> one b128 LDS store.
// mode 0: input += pbias[c] (in-conv).  mode 1: fused GroupNorm + SiLU.
// ---------------------------------------------------------------------------
__global__ __launch_bounds__(128)
void conv3d_wmma_kernel(const float* __restrict__ in,      // [16][Cin][32768]
                        const __bf16* __restrict__ Wt,     // [64][Cin*27] bf16, tap-major
                        const float* __restrict__ bias,    // [64]
                        const float* residual,             // nullable, [16][64][32768]
                        float* out,                        // [16][64][32768]
                        int Cin, int kShift, int mode,
                        const float* __restrict__ stats,   // [16][8][2] (mode 1)
                        const float* __restrict__ gamma,   // [Cin] (mode 1)
                        const float* __restrict__ beta,    // [Cin] (mode 1)
                        const float* __restrict__ pbias)   // [Cin] (mode 0)
{
  __shared__ alignas(16) __bf16 BsT[32][40];   // [n][k] im2col tile, padded rows
  __shared__ float2 chs[NCH];                  // per-channel (scale, shift)
  __shared__ int2   tbl[27];                   // (wrapped row offset, dx) per tap
#if USE_TDM
  __shared__ alignas(16) __bf16 As[64 * 40];   // A tile: 64 rows x 32 k, 80B rows
#endif

  const int tid  = threadIdx.x;
  const int wave = tid >> 5;
  const int lane = tid & 31;

  const int blk = blockIdx.x;                  // 16 * 1024 blocks
  const int b   = blk >> 10;
  const int zy  = blk & 1023;
  const int y   = zy & 31;
  const int z   = zy >> 5;

  for (int c = tid; c < Cin; c += 128) {
    if (mode == 1) {
      int g = c >> 3;
      float mu = stats[(b * 8 + g) * 2 + 0], rstd = stats[(b * 8 + g) * 2 + 1];
      float sc = rstd * gamma[c];
      chs[c] = make_float2(sc, beta[c] - mu * sc);
    } else {
      chs[c] = make_float2(1.f, pbias[c]);
    }
  }
  if (tid < 27) {
    int dz = tid / 9 - 1;
    int r9 = tid % 9;
    int dy = r9 / 3 - 1, dx = r9 % 3 - 1;
    tbl[tid] = make_int2(((z + dz) & 31) * 1024 + ((y + dy) & 31) * 32, dx);
  }
  __syncthreads();

  const long long inB  = (long long)b * Cin * SPAT;
  const int Ktot  = Cin * 27;
  const int cMask = Cin - 1;
  const int mbase = wave << 4;
  const int m     = lane & 15;
  const int half  = lane >> 4;
  const int gn    = tid & 31;                  // gather: fixed n per thread
  const int kkb   = (tid >> 5) << 3;           // gather: contiguous 8-k run

#if USE_TDM
  // Loop-invariant D# group 1: 2B elements, 2D tensor Ktot x 64, tile 32 x 64,
  // LDS padding: 4 DWORDs after every 16 DWORDs (64B row -> 80B row).
  i32x8 g1;
  g1[0] = 0x00010000 | (1 << 20) | (3 << 22) | (3 << 25);  // data_size|pad_en|intv|amt
  g1[1] = (int)((Ktot & 0xFFFF) << 16);                    // tensor_dim0[15:0]
  g1[2] = (Ktot >> 16) | (64 << 16);                       // dim0 hi | tensor_dim1 lo
  g1[3] = (32 << 16);                                      // tensor_dim1 hi | tile_dim0
  g1[4] = 64;                                              // tile_dim1 | tile_dim2=0
  g1[5] = Ktot;                                            // tensor_dim0_stride lo32
  g1[6] = 0;
  g1[7] = 0;
  i32x4 gz = {0, 0, 0, 0};
#if TDM_ARGS6
  i32x8 z8 = {0, 0, 0, 0, 0, 0, 0, 0};
#endif
  const unsigned long long wbase = (unsigned long long)(uintptr_t)Wt;
  const unsigned ldsA = (unsigned)(uintptr_t)(&As[0]);
#endif

  v8f acc0 = {0.f, 0.f, 0.f, 0.f, 0.f, 0.f, 0.f, 0.f};
  v8f acc1 = {0.f, 0.f, 0.f, 0.f, 0.f, 0.f, 0.f, 0.f};

  for (int kc = 0; kc < Ktot; kc += 32) {
#if USE_TDM
    if (wave == 0) {
      unsigned long long ga = wbase + (unsigned long long)(unsigned)(kc << 1);
      u32x4 g0;
      g0.x = 1u;                                           // count=1
      g0.y = ldsA;                                         // lds_addr
      g0.z = (unsigned)ga;                                 // global_addr lo
      g0.w = ((unsigned)(ga >> 32) & 0x01FFFFFFu) | 0x80000000u;  // hi | type=2
#if TDM_ARGS6
      __builtin_amdgcn_tensor_load_to_lds(g0, g1, gz, gz, z8, 0);
#else
      __builtin_amdgcn_tensor_load_to_lds(g0, g1, gz, gz, 0);
#endif
    }
#endif
    // Cooperative im2col gather (fused affine + SiLU); tap-major K => shift/mask.
    // Thread: fixed n, 8 contiguous k -> batched loads + single b128 LDS store.
    {
      v8bf bv;
#pragma unroll
      for (int j = 0; j < 8; ++j) {
        int k   = kc + kkb + j;
        int tap = k >> kShift;
        int c   = k & cMask;
        int2 tt = tbl[tap];
        int px  = (gn + tt.y) & 31;
        float v = in[inB + (long long)c * SPAT + tt.x + px];
        float2 sb = chs[c];
        v = v * sb.x + sb.y;
        if (mode == 1) v = silu_fast(v);
        bv[j] = (__bf16)v;
      }
      *(v8bf*)(&BsT[gn][kkb]) = bv;
    }
#if USE_TDM
    if (wave == 0) __builtin_amdgcn_s_wait_tensorcnt(0);
#endif
    __syncthreads();

    // A fragment per ISA 16-bit A 16x32 lane layout.
    union { v16bf v; v8bf h[2]; } A, B0, B1;
#if USE_TDM
    const __bf16* arow = &As[(mbase + m) * 40];
#else
    const __bf16* arow = Wt + (mbase + m) * Ktot + kc;
    __builtin_prefetch((const void*)(arow + 32), 0, 1);
#endif
    A.h[0] = *(const v8bf*)(arow + half * 8);
    A.h[1] = *(const v8bf*)(arow + 16 + half * 8);
    // Two B fragments (x 0..15 and 16..31) sharing A.
    const __bf16* brow0 = &BsT[m][half * 8];
    const __bf16* brow1 = &BsT[m + 16][half * 8];
    B0.h[0] = *(const v8bf*)(brow0);
    B0.h[1] = *(const v8bf*)(brow0 + 16);
    B1.h[0] = *(const v8bf*)(brow1);
    B1.h[1] = *(const v8bf*)(brow1 + 16);

    acc0 = __builtin_amdgcn_wmma_f32_16x16x32_bf16(false, A.v, false, B0.v,
                                                   (short)0, acc0, false, false);
    acc1 = __builtin_amdgcn_wmma_f32_16x16x32_bf16(false, A.v, false, B1.v,
                                                   (short)0, acc1, false, false);
    __syncthreads();
  }

  // Writeout per 16x16 f32 C/D layout: lane = N, VGPR r = M row (+8 for half 1).
  const int n = lane & 15;
#pragma unroll
  for (int r = 0; r < 8; ++r) {
    int o = mbase + (half << 3) + r;
    long long obase = (long long)b * NCH * SPAT + (long long)o * SPAT
                    + z * 1024 + y * 32;
    float v0 = acc0[r] + bias[o];
    float v1 = acc1[r] + bias[o];
    if (residual) { v0 += residual[obase + n]; v1 += residual[obase + 16 + n]; }
    out[obase + n]      = v0;
    out[obase + 16 + n] = v1;
  }
}

// ---------------------------------------------------------------------------
__global__ __launch_bounds__(256)
void head_kernel(const float* __restrict__ x,
                 const float* __restrict__ lsw, const float* __restrict__ lsb,
                 const float* __restrict__ tw,  const float* __restrict__ tb,
                 const float* __restrict__ phi_free,
                 float* __restrict__ phi_im, float* __restrict__ log_det) {
  __shared__ float Wls[32 * 64], Wta[32 * 64];
  for (int i = threadIdx.x; i < 2048; i += 256) { Wls[i] = lsw[i]; Wta[i] = tw[i]; }
  __syncthreads();
  int b = blockIdx.x >> 7;                     // 128 blocks per batch
  int s = ((blockIdx.x & 127) << 8) + threadIdx.x;
  const float* xb = x + (long long)b * NCH * SPAT + s;
  float xv[64];
#pragma unroll
  for (int c = 0; c < 64; ++c) xv[c] = xb[(long long)c * SPAT];
  float lsum = 0.f;
  for (int o = 0; o < 32; ++o) {
    float a0 = lsb[o], a1 = tb[o];
#pragma unroll
    for (int c = 0; c < 64; ++c) { a0 += Wls[o * 64 + c] * xv[c]; a1 += Wta[o * 64 + c] * xv[c]; }
    float ls = tanh_fast(a0) * 0.3f;
    lsum += ls;
    long long pidx = (long long)b * VOL + (long long)o * SPAT + s;
    phi_im[pidx] = __expf(ls) * phi_free[pidx] + a1;
  }
  __shared__ float red[256];
  red[threadIdx.x] = lsum; __syncthreads();
  for (int st = 128; st > 0; st >>= 1) {
    if (threadIdx.x < st) red[threadIdx.x] += red[threadIdx.x + st];
    __syncthreads();
  }
  if (threadIdx.x == 0) atomicAdd(log_det + b, red[0]);
}

// ---------------------------------------------------------------------------
extern "C" void kernel_launch(void* const* d_in, const int* in_sizes, int n_in,
                              void* d_out, int out_size, void* d_ws, size_t ws_size,
                              hipStream_t stream) {
  (void)in_sizes; (void)n_in; (void)out_size; (void)ws_size;

  const float* z      = (const float*)d_in[0];
  const float* p_vec  = (const float*)d_in[1];
  const float* pe_w1  = (const float*)d_in[2];
  const float* pe_b1  = (const float*)d_in[3];
  const float* pe_w2  = (const float*)d_in[4];
  const float* pe_b2  = (const float*)d_in[5];
  const float* pe_w3  = (const float*)d_in[6];
  const float* pe_b3  = (const float*)d_in[7];
  const float* pp_w   = (const float*)d_in[8];
  const float* pp_b   = (const float*)d_in[9];
  const float* in_w   = (const float*)d_in[10];
  const float* in_b   = (const float*)d_in[11];
  const float* r1_g1  = (const float*)d_in[12];
  const float* r1_b1  = (const float*)d_in[13];
  const float* r1_cw1 = (const float*)d_in[14];
  const float* r1_cb1 = (const float*)d_in[15];
  const float* r1_g2  = (const float*)d_in[16];
  const float* r1_b2  = (const float*)d_in[17];
  const float* r1_cw2 = (const float*)d_in[18];
  const float* r1_cb2 = (const float*)d_in[19];
  const float* r2_g1  = (const float*)d_in[20];
  const float* r2_b1  = (const float*)d_in[21];
  const float* r2_cw1 = (const float*)d_in[22];
  const float* r2_cb1 = (const float*)d_in[23];
  const float* r2_g2  = (const float*)d_in[24];
  const float* r2_b2  = (const float*)d_in[25];
  const float* r2_cw2 = (const float*)d_in[26];
  const float* r2_cb2 = (const float*)d_in[27];
  const float* ls_w   = (const float*)d_in[28];
  const float* ls_b   = (const float*)d_in[29];
  const float* tau_w  = (const float*)d_in[30];
  const float* tau_b  = (const float*)d_in[31];

  float* phi  = (float*)d_out;                 // phi_free [16M]
  float* phiI = phi + (long long)NB * VOL;     // phi_im   [16M]
  float* ldet = phi + 2ll * NB * VOL;          // log_det  [16]

  // Workspace layout (imag aliases X: FFT phase ends before convs begin).
  char* ws = (char*)d_ws;
  float*  IM    = (float*)ws;                                   //  64 MB (FFT imag)
  float*  X     = (float*)ws;                                   // 128 MB activations
  float*  Y     = (float*)(ws + (1ull << 27));                  // 128 MB activations
  __bf16* Wt_in = (__bf16*)(ws + (1ull << 28));
  __bf16* Wt_1a = (__bf16*)(ws + (1ull << 28) + 1ull * 262144);
  __bf16* Wt_1b = (__bf16*)(ws + (1ull << 28) + 2ull * 262144);
  __bf16* Wt_2a = (__bf16*)(ws + (1ull << 28) + 3ull * 262144);
  __bf16* Wt_2b = (__bf16*)(ws + (1ull << 28) + 4ull * 262144);
  float*  ST    = (float*)(ws + (1ull << 28) + 5ull * 262144);  // GN stats [16][8][2]
  float*  PB    = ST + 1024;                                    // p_bias [32]
  float*  SC    = PB + 64;                                      // log-det scalar

  // --- Spectral preconditioning: 4 fwd FFT passes, scale, 4 inv passes ------
  fft_pass_kernel<<<2048, 256, 0, stream>>>(z, phi, IM, 15, 1, -1.f);
  fft_pass_kernel<<<2048, 256, 0, stream>>>(nullptr, phi, IM, 10, 0, -1.f);
  fft_pass_kernel<<<2048, 256, 0, stream>>>(nullptr, phi, IM, 5, 0, -1.f);
  fft_pass_kernel<<<2048, 256, 0, stream>>>(nullptr, phi, IM, 0, 0, -1.f);
  spectral_scale_kernel<<<65536, 256, 0, stream>>>(phi, IM, p_vec);
  fft_pass_kernel<<<2048, 256, 0, stream>>>(nullptr, phi, IM, 15, 0, 1.f);
  fft_pass_kernel<<<2048, 256, 0, stream>>>(nullptr, phi, IM, 10, 0, 1.f);
  fft_pass_kernel<<<2048, 256, 0, stream>>>(nullptr, phi, IM, 5, 0, 1.f);
  fft_pass_kernel<<<2048, 256, 0, stream>>>(nullptr, phi, IM, 0, 0, 1.f);
  // phi now holds phi_free.

  // --- Param MLP + weight re-layout (tap-major bf16) -----------------------
  param_mlp_kernel<<<1, 32, 0, stream>>>(p_vec, pe_w1, pe_b1, pe_w2, pe_b2,
                                         pe_w3, pe_b3, pp_w, pp_b, PB);
  prep_weights_kernel<<<(55296 + 255) / 256, 256, 0, stream>>>(in_w, Wt_in, 32);
  prep_weights_kernel<<<(110592 + 255) / 256, 256, 0, stream>>>(r1_cw1, Wt_1a, 64);
  prep_weights_kernel<<<(110592 + 255) / 256, 256, 0, stream>>>(r1_cw2, Wt_1b, 64);
  prep_weights_kernel<<<(110592 + 255) / 256, 256, 0, stream>>>(r2_cw1, Wt_2a, 64);
  prep_weights_kernel<<<(110592 + 255) / 256, 256, 0, stream>>>(r2_cw2, Wt_2b, 64);

  // --- log_det constant: sum(-0.5*log(kernel)) broadcast to all batches ----
  zero_kernel<<<1, 32, 0, stream>>>(SC, 1);
  logdet_const_kernel<<<1024, 256, 0, stream>>>(p_vec, SC);
  logdet_bcast_kernel<<<1, 32, 0, stream>>>(SC, ldet);

  // --- ResNet conditioner: 5 WMMA convs + 4 GN stats -----------------------
  conv3d_wmma_kernel<<<16384, 128, 0, stream>>>(phi, Wt_in, in_b, nullptr, X,
                                                32, 5, 0, nullptr, nullptr, nullptr, PB);
  gn_stats_kernel<<<128, 256, 0, stream>>>(X, ST);
  conv3d_wmma_kernel<<<16384, 128, 0, stream>>>(X, Wt_1a, r1_cb1, nullptr, Y,
                                                64, 6, 1, ST, r1_g1, r1_b1, nullptr);
  gn_stats_kernel<<<128, 256, 0, stream>>>(Y, ST);
  conv3d_wmma_kernel<<<16384, 128, 0, stream>>>(Y, Wt_1b, r1_cb2, X, X,
                                                64, 6, 1, ST, r1_g2, r1_b2, nullptr);
  gn_stats_kernel<<<128, 256, 0, stream>>>(X, ST);
  conv3d_wmma_kernel<<<16384, 128, 0, stream>>>(X, Wt_2a, r2_cb1, nullptr, Y,
                                                64, 6, 1, ST, r2_g1, r2_b1, nullptr);
  gn_stats_kernel<<<128, 256, 0, stream>>>(Y, ST);
  conv3d_wmma_kernel<<<16384, 128, 0, stream>>>(Y, Wt_2b, r2_cb2, X, X,
                                                64, 6, 1, ST, r2_g2, r2_b2, nullptr);

  // --- Heads + imaginary layer + per-batch log_det sums --------------------
  head_kernel<<<2048, 256, 0, stream>>>(X, ls_w, ls_b, tau_w, tau_b,
                                        phi, phiI, ldet);
}
